// MoCo_60464549593470
// MI455X (gfx1250) — matrix operand order repaired and checked by reading, not compiled, not run on hardware
//
#include <hip/hip_runtime.h>
#include <hip/hip_bf16.h>

typedef __attribute__((ext_vector_type(16))) __bf16 v16bf;
typedef __attribute__((ext_vector_type(8)))  __bf16 v8bf;
typedef __attribute__((ext_vector_type(8)))  float  v8f;
typedef unsigned long long u64;

#define B_ROWS   4096
#define D_DIM    1024
#define K_FULL   131072
#define STRIDE   10
#define KP       13108        // ceil(131072/10)
#define T_LAB    50
#define NUM_POS  16
#define INV_SQRT2 0.70710678f

// ---------------------------------------------------------------------------
// 1) Row L2-normalize (with +1e-8 like reference) and convert to bf16.
//    srcStride lets the same kernel do the [::10] gather for the queue.
// ---------------------------------------------------------------------------
__global__ void norm_rows_bf16(const float* __restrict__ src,
                               __bf16* __restrict__ dst,
                               int srcStride) {
    const int row = blockIdx.x;
    const int tid = threadIdx.x;
    const float* s = src + (long)row * srcStride * D_DIM;
    __shared__ float red[256];

    float ss = 0.f;
    for (int j = tid; j < D_DIM; j += 256) { float v = s[j]; ss += v * v; }
    red[tid] = ss;
    __syncthreads();
    for (int o = 128; o > 0; o >>= 1) {
        if (tid < o) red[tid] += red[tid + o];
        __syncthreads();
    }
    const float scale = 1.0f / (sqrtf(red[0]) + 1e-8f);
    __bf16* d = dst + (long)row * D_DIM;
    for (int j = tid; j < D_DIM; j += 256)
        d[j] = (__bf16)(s[j] * scale);
}

// ---------------------------------------------------------------------------
// 2) Pack the 50 (l=0,l=1) label bits of each row into two 64-bit masks.
// ---------------------------------------------------------------------------
__global__ void pack_labels(const int* __restrict__ lab,
                            u64* __restrict__ L0, u64* __restrict__ L1,
                            int srcStride, int nrows) {
    const int r = blockIdx.x * blockDim.x + threadIdx.x;
    if (r >= nrows) return;
    const int* p = lab + (long)r * srcStride * (T_LAB * 2);
    u64 b0 = 0, b1 = 0;
    #pragma unroll 10
    for (int t = 0; t < T_LAB; ++t) {
        if (p[2 * t])     b0 |= (1ull << t);
        if (p[2 * t + 1]) b1 |= (1ull << t);
    }
    L0[r] = b0; L1[r] = b1;
}

// ---------------------------------------------------------------------------
// 3) One wave32 per anchor row: exact label-sim mask via popcount, ballot,
//    keep the first <=16 masked columns (cumsum/rank semantics), early exit.
// ---------------------------------------------------------------------------
__global__ void select_cols(const u64* __restrict__ A0, const u64* __restrict__ A1,
                            const u64* __restrict__ Q0, const u64* __restrict__ Q1,
                            int* __restrict__ sel, int* __restrict__ cnt) {
    const int b = blockIdx.x;
    const int lane = threadIdx.x;      // 0..31, wave32
    if (lane < NUM_POS) sel[b * NUM_POS + lane] = 0;   // safe default row
    const u64 a0 = A0[b], a1 = A1[b];
    int total = 0;
    for (int base = 0; base < KP && total < NUM_POS; base += 32) {
        const int k = base + lane;
        bool m = false;
        if (k < KP) {
            const u64 q0 = Q0[k], q1 = Q1[k];
            // per-t dot is 1 for identical nonzero codes, 1/sqrt2 for (1,1) vs single
            const u64 one  = (a0 & ~a1 & q0 & ~q1) | (~a0 & a1 & ~q0 & q1) | (a0 & a1 & q0 & q1);
            const u64 half = ((a0 & a1) & (q0 ^ q1)) | ((q0 & q1) & (a0 ^ a1));
            const float sim50 = (float)__popcll(one) + INV_SQRT2 * (float)__popcll(half);
            m = (sim50 >= 25.0f);      // lab_sim = sim50/50 >= 0.5
        }
        const u64 bal = __ballot(m);
        const int pre = __popcll(bal & ((1ull << lane) - 1ull));
        if (m && (total + pre) < NUM_POS) sel[b * NUM_POS + total + pre] = k;
        total += __popcll(bal);
    }
    if (lane == 0) cnt[b] = (total < NUM_POS) ? total : NUM_POS;
}

// ---------------------------------------------------------------------------
// 4) One wave32 per anchor row: 16x16x1024 bf16 WMMA GEMM between the aligned
//    16-row q tile and this row's <=16 selected queue rows; epilogue applies
//    softplus(-sims/T) on the single relevant output row and reduces.
// ---------------------------------------------------------------------------
__global__ void wmma_loss(const __bf16* __restrict__ Qn,
                          const __bf16* __restrict__ Kn,
                          const int* __restrict__ sel,
                          const int* __restrict__ cnt,
                          float* __restrict__ per_anchor) {
    const int b    = blockIdx.x;
    const int lane = threadIdx.x;          // 0..31
    const int b0   = b & ~15;              // aligned 16-row A tile
    const int c    = cnt[b];

    // Per-lane source rows; A and B share contiguous k order k = 16*(lane/16) + s.
    const int  rowA  = b0 + (lane & 15);
    const int  kbA   = (lane < 16) ? 0 : 16;   // A: lanes 0-15 K 0..15; lanes 16-31 K 16..31
    const long rowB  = sel[b * NUM_POS + (lane & 15)];
    const int  kbB   = (lane < 16) ? 0 : 16;   // B: lanes 0-15 K 0..15; lanes 16-31 K 16..31
    const __bf16* aRow = Qn + (long)rowA * D_DIM;
    const __bf16* bRow = Kn + rowB * D_DIM;

    v8f acc = {};
    for (int kk = 0; kk < D_DIM; kk += 32) {
        // A fragment: two adjacent 8-half (16B) chunks = 16 contiguous K
        v8bf lo = *(const v8bf*)(aRow + kk + kbA);
        v8bf hi = *(const v8bf*)(aRow + kk + kbA + 8);
        v16bf afrag;
        #pragma unroll
        for (int j = 0; j < 8; ++j) { afrag[j] = lo[j]; afrag[8 + j] = hi[j]; }
        // B fragment: one contiguous 16-half (32B) chunk
        v16bf bfrag = *(const v16bf*)(bRow + kk + kbB);

        acc = __builtin_amdgcn_wmma_f32_16x16x32_bf16(
            false, afrag, false, bfrag, (short)0, acc, false, false);
    }

    // Extract output row m = b & 15: VGPR m%8, lanes 0-15 if m<8 else 16-31.
    const int  i        = b & 15;
    const bool ownerLow = (i < 8);
    const int  r        = ownerLow ? i : (i - 8);
    float v = acc[0];
    #pragma unroll
    for (int t = 1; t < 8; ++t) if (r == t) v = acc[t];

    const bool owner = ownerLow ? (lane < 16) : (lane >= 16);
    const int  j     = lane & 15;
    float contrib = 0.f;
    if (owner && j < c) {
        const float s = 2.0f * v;                    // sims = dot / TEMPERATURE(0.5)
        const float z = -s;                          // loss = softplus(-sims), stable
        contrib = fmaxf(z, 0.f) + log1pf(expf(-fabsf(z)));
    }
    #pragma unroll
    for (int off = 16; off > 0; off >>= 1)
        contrib += __shfl_xor(contrib, off, 32);
    if (lane == 0)
        per_anchor[b] = (c > 0) ? (contrib / (float)c) : 0.f;
}

// ---------------------------------------------------------------------------
// 5) Mean over the 4096 per-anchor losses -> scalar output.
// ---------------------------------------------------------------------------
__global__ void final_mean(const float* __restrict__ per_anchor, float* __restrict__ out) {
    __shared__ float red[256];
    const int tid = threadIdx.x;
    float s = 0.f;
    for (int i = tid; i < B_ROWS; i += 256) s += per_anchor[i];
    red[tid] = s;
    __syncthreads();
    for (int o = 128; o > 0; o >>= 1) {
        if (tid < o) red[tid] += red[tid + o];
        __syncthreads();
    }
    if (tid == 0) out[0] = red[0] / (float)B_ROWS;
}

// ---------------------------------------------------------------------------
static inline size_t align256(size_t x) { return (x + 255) & ~(size_t)255; }

extern "C" void kernel_launch(void* const* d_in, const int* in_sizes, int n_in,
                              void* d_out, int out_size, void* d_ws, size_t ws_size,
                              hipStream_t stream) {
    (void)in_sizes; (void)n_in; (void)out_size; (void)ws_size;
    const float* x_q            = (const float*)d_in[0];
    const int*   x_label        = (const int*)d_in[1];
    const float* queue_features = (const float*)d_in[2];
    const int*   queue_labels   = (const int*)d_in[3];
    float*       out            = (float*)d_out;

    char* ws = (char*)d_ws;
    size_t off = 0;
    __bf16* Qn = (__bf16*)(ws + off); off = align256(off + (size_t)B_ROWS * D_DIM * 2);
    __bf16* Kn = (__bf16*)(ws + off); off = align256(off + (size_t)KP * D_DIM * 2);
    u64* A0 = (u64*)(ws + off); off = align256(off + (size_t)B_ROWS * 8);
    u64* A1 = (u64*)(ws + off); off = align256(off + (size_t)B_ROWS * 8);
    u64* Q0 = (u64*)(ws + off); off = align256(off + (size_t)KP * 8);
    u64* Q1 = (u64*)(ws + off); off = align256(off + (size_t)KP * 8);
    int* sel = (int*)(ws + off); off = align256(off + (size_t)B_ROWS * NUM_POS * 4);
    int* cnt = (int*)(ws + off); off = align256(off + (size_t)B_ROWS * 4);
    float* per_anchor = (float*)(ws + off); off = align256(off + (size_t)B_ROWS * 4);

    norm_rows_bf16<<<B_ROWS, 256, 0, stream>>>(x_q, Qn, 1);
    norm_rows_bf16<<<KP, 256, 0, stream>>>(queue_features, Kn, STRIDE);
    pack_labels<<<(B_ROWS + 255) / 256, 256, 0, stream>>>(x_label, A0, A1, 1, B_ROWS);
    pack_labels<<<(KP + 255) / 256, 256, 0, stream>>>(queue_labels, Q0, Q1, STRIDE, KP);
    select_cols<<<B_ROWS, 32, 0, stream>>>(A0, A1, Q0, Q1, sel, cnt);
    wmma_loss<<<B_ROWS, 32, 0, stream>>>(Qn, Kn, sel, cnt, per_anchor);
    final_mean<<<1, 256, 0, stream>>>(per_anchor, out);
}
